// Attention_3315714753146
// MI455X (gfx1250) — matrix operand order repaired and checked by reading, not compiled, not run on hardware
//
#include <hip/hip_runtime.h>
#include <hip/hip_bf16.h>

typedef __attribute__((ext_vector_type(16))) __bf16 v16bf;
typedef __attribute__((ext_vector_type(8)))  __bf16 v8bf;
typedef __attribute__((ext_vector_type(8)))  float  v8f;

#define HEADS  8
#define DHEAD  64
#define BATCH  4
#define SEQ    1024
#define DMODEL 512
#define INNER  512          // HEADS*DHEAD
#define QKVC   1536
#define LA_C   0.33f
#define LD_C   0.33f
#define LG_C   0.33f
#define BIGNEG 1.0e38f      // large-but-bf16-safe mask value (avoids inf-inf NaN)

// smem: 8 waves * (16 x 1024) bf16 score tiles + 1KB mask row + 8*16 f32 rowmax
#define SMEM_BYTES (8*16*SEQ*2 + 1024 + 8*16*4)

#define WMMA_BF16(a,b,c) \
  __builtin_amdgcn_wmma_f32_16x16x32_bf16(false,(a),false,(b),(short)0,(c),false,false)

// Build a 16-half WMMA fragment from two 16-byte chunks.
// A-frag: p1 = p0 + 8 halves.  B-frag: p1 = p0 + 8 halves.
__device__ __forceinline__ v16bf ld_frag(const __bf16* p0, const __bf16* p1) {
  v8bf lo = *(const v8bf*)p0;
  v8bf hi = *(const v8bf*)p1;
  return __builtin_shufflevector(lo, hi, 0,1,2,3,4,5,6,7,8,9,10,11,12,13,14,15);
}

// One K-step worth of fragments for a 16x64 output tile (1 A-frag + 4 B-frags)
struct FragSet { v16bf a; v16bf b0, b1, b2, b3; };

__device__ __forceinline__ FragSet load_set(const __bf16* arow,
                                            const __bf16* b0, const __bf16* b1,
                                            const __bf16* b2, const __bf16* b3,
                                            int k0) {
  FragSet f;
  f.a  = ld_frag(arow + k0, arow + k0 + 8);
  f.b0 = ld_frag(b0 + k0, b0 + k0 + 8);
  f.b1 = ld_frag(b1 + k0, b1 + k0 + 8);
  f.b2 = ld_frag(b2 + k0, b2 + k0 + 8);
  f.b3 = ld_frag(b3 + k0, b3 + k0 + 8);
  return f;
}

__device__ __forceinline__ void mma_set(v8f acc[4], const FragSet& f) {
  acc[0] = WMMA_BF16(f.a, f.b0, acc[0]);
  acc[1] = WMMA_BF16(f.a, f.b1, acc[1]);
  acc[2] = WMMA_BF16(f.a, f.b2, acc[2]);
  acc[3] = WMMA_BF16(f.a, f.b3, acc[3]);
}

// ---------------------------------------------------------------- conversions
__global__ __launch_bounds__(256) void cvt_kernel(
    const float* __restrict__ x, const float* __restrict__ wqkv,
    const float* __restrict__ wout,
    __bf16* __restrict__ Xb, __bf16* __restrict__ WqkvT, __bf16* __restrict__ WoutT) {
  int i = blockIdx.x * 256 + threadIdx.x;
  const int NX  = BATCH*SEQ*DMODEL;   // 2097152
  const int NW1 = DMODEL*QKVC;        // 786432
  const int NW2 = DMODEL*INNER;       // 262144
  if (i < NX) { Xb[i] = (__bf16)x[i]; return; }
  i -= NX;
  if (i < NW1) { int k = i / QKVC, c = i % QKVC;
                 WqkvT[(size_t)c*DMODEL + k] = (__bf16)wqkv[i]; return; }
  i -= NW1;
  if (i < NW2) { int k = i / INNER, c = i % INNER;
                 WoutT[(size_t)c*DMODEL + k] = (__bf16)wout[i]; }
}

// ---------------------------------------------------------------- QKV GEMM
// [4096,512] @ [512,1536] -> scatter into Q (pre-scaled), K, V^T (bf16)
__global__ __launch_bounds__(256) void qkv_gemm_kernel(
    const __bf16* __restrict__ Xb, const __bf16* __restrict__ WqkvT,
    __bf16* __restrict__ Qb, __bf16* __restrict__ Kb, __bf16* __restrict__ VTb) {
  const int lane = threadIdx.x & 31, wave = threadIdx.x >> 5;
  const int lr = lane & 15, lh = lane >> 4;
  const int row0 = blockIdx.x * 128 + wave * 16;
  const int cb   = blockIdx.y;              // 64-wide column block: aligns to (h,t)
  const int col0 = cb * 64;
  const int h = cb / 3, t = cb % 3;

  v8f acc[4] = {};
  const __bf16* arow = Xb + (size_t)(row0 + lr)*DMODEL + lh*16;
  const __bf16* bp0 = WqkvT + (size_t)(col0 +  0 + lr)*DMODEL + lh*16;
  const __bf16* bp1 = WqkvT + (size_t)(col0 + 16 + lr)*DMODEL + lh*16;
  const __bf16* bp2 = WqkvT + (size_t)(col0 + 32 + lr)*DMODEL + lh*16;
  const __bf16* bp3 = WqkvT + (size_t)(col0 + 48 + lr)*DMODEL + lh*16;

  // software pipeline: next K-step's loads stay in flight across current WMMAs
  FragSet cur = load_set(arow, bp0, bp1, bp2, bp3, 0);
  for (int k0 = 0; k0 < DMODEL - 32; k0 += 32) {
    FragSet nxt = load_set(arow, bp0, bp1, bp2, bp3, k0 + 32);
    mma_set(acc, cur);
    cur = nxt;
  }
  mma_set(acc, cur);

  #pragma unroll
  for (int f = 0; f < 4; ++f) {
    #pragma unroll
    for (int r = 0; r < 8; ++r) {
      const int grow = row0 + lh*8 + r;
      const int d = f*16 + lr;
      const int b = grow >> 10, n = grow & (SEQ-1);
      const float v = acc[f][r];
      const size_t bh = (size_t)(b*HEADS + h);
      if (t == 0)      Qb[(bh*SEQ + n)*DHEAD + d] = (__bf16)(v * 0.125f); // *Dh^-0.5
      else if (t == 1) Kb[(bh*SEQ + n)*DHEAD + d] = (__bf16)v;
      else             VTb[(bh*DHEAD + d)*SEQ + n] = (__bf16)v;           // transposed
    }
  }
}

// ---------------------------------------------------------------- attention
__global__ __launch_bounds__(256) void attn_kernel(
    const unsigned char* __restrict__ mask,
    const float* __restrict__ adj, const float* __restrict__ dist,
    const __bf16* __restrict__ Qb, const __bf16* __restrict__ Kb,
    const __bf16* __restrict__ VTb, __bf16* __restrict__ Ob) {
  extern __shared__ char smem[];
  const int lane = threadIdx.x & 31, wave = threadIdx.x >> 5;
  const int lr = lane & 15, lh = lane >> 4;
  int idx = blockIdx.x;
  const int ib = idx & 7; idx >>= 3;
  const int h  = idx & 7; const int b = idx >> 3;
  const int i0 = ib*128 + wave*16;

  __bf16* P = (__bf16*)smem + (size_t)wave * (16*SEQ);
  unsigned char* mLds = (unsigned char*)(smem + 8*16*SEQ*2);
  float* rmLds = (float*)(smem + 8*16*SEQ*2 + 1024) + wave*16;

  // stage this batch's mask row (1024 bytes) into LDS
  ((uint32_t*)mLds)[threadIdx.x] = ((const uint32_t*)(mask + (size_t)b*SEQ))[threadIdx.x];
  __syncthreads();

  const size_t bh = (size_t)(b*HEADS + h);
  const __bf16* qrow = Qb + (bh*SEQ + i0 + lr)*DHEAD;
  const v16bf aq0 = ld_frag(qrow + lh*16,      qrow + lh*16 + 8);       // d 0..31
  const v16bf aq1 = ld_frag(qrow + 32 + lh*16, qrow + 32 + lh*16 + 8);  // d 32..63

  unsigned miMask = 0;
  #pragma unroll
  for (int r = 0; r < 8; ++r)
    miMask |= (unsigned)(mLds[i0 + lh*8 + r] != 0) << r;

  float rm[8];
  #pragma unroll
  for (int r = 0; r < 8; ++r) rm[r] = -BIGNEG;

  // ---- pass 1: S = (q*scale) k^T, masked, bf16 scores -> LDS; row max in regs
  const __bf16* kbase = Kb + bh*SEQ*DHEAD;
  auto ldk = [&](int j0, v16bf& b0, v16bf& b1) {
    const __bf16* krow = kbase + (size_t)(j0 + lr)*DHEAD + lh*16;
    b0 = ld_frag(krow,      krow + 8);    // d 0..31
    b1 = ld_frag(krow + 32, krow + 40);   // d 32..63
  };
  auto qk_body = [&](int j0, v16bf b0, v16bf b1) {
    v8f s = {};
    s = WMMA_BF16(aq0, b0, s);
    s = WMMA_BF16(aq1, b1, s);
    const unsigned mj = mLds[j0 + lr];
    #pragma unroll
    for (int r = 0; r < 8; ++r) {
      const float sv = (mj && ((miMask >> r) & 1)) ? s[r] : -BIGNEG;
      rm[r] = fmaxf(rm[r], sv);
      P[(lh*8 + r)*SEQ + j0 + lr] = (__bf16)sv;
    }
  };
  v16bf kb0, kb1;
  ldk(0, kb0, kb1);
  for (int j0 = 0; j0 < SEQ - 16; j0 += 16) {
    v16bf nb0, nb1;
    ldk(j0 + 16, nb0, nb1);            // in flight across current WMMAs
    qk_body(j0, kb0, kb1);
    kb0 = nb0; kb1 = nb1;
  }
  qk_body(SEQ - 16, kb0, kb1);

  // reduce row-max across the 16-lane C-fragment groups (stays inside each half)
  #pragma unroll
  for (int m = 1; m < 16; m <<= 1) {
    #pragma unroll
    for (int r = 0; r < 8; ++r) rm[r] = fmaxf(rm[r], __shfl_xor(rm[r], m, 32));
  }
  // lanes 0 / 16 publish row maxima for rows 0-7 / 8-15 of this wave's tile
  if (lr == 0) {
    #pragma unroll
    for (int r = 0; r < 8; ++r) rmLds[lh*8 + r] = rm[r];
  }
  __syncthreads();

  // ---- pass 2: softmax + LA scale + masked LG*adj + LD*exp(-dist), in place
  {
    const int   jb   = lh * 512;          // this lane's half-row
    const float mrow = rmLds[lr];
    __bf16* prow = P + lr*SEQ + jb;
    float sum = 0.f;
    for (int j = 0; j < 512; ++j) {
      const float e = __expf((float)prow[j] - mrow);
      prow[j] = (__bf16)e;
      sum += e;
    }
    sum += __shfl_xor(sum, 16, 32);       // combine the two half-rows
    const float sc = LA_C / sum;
    const int gi = i0 + lr;
    const unsigned mi = mLds[gi];
    const float* arow = adj  + ((size_t)b*SEQ + gi)*SEQ + jb;
    const float* drow = dist + ((size_t)b*SEQ + gi)*SEQ + jb;
    for (int j = 0; j < 512; j += 4) {
      __builtin_prefetch(arow + j + 128, 0, 1);   // global_prefetch_b8
      __builtin_prefetch(drow + j + 128, 0, 1);
      const float4 av = *(const float4*)(arow + j);
      const float4 dv = *(const float4*)(drow + j);
      #pragma unroll
      for (int u = 0; u < 4; ++u) {
        float e = (float)prow[j + u] * sc;
        if (mi && mLds[jb + j + u])
          e += LG_C * (&av.x)[u] + LD_C * __expf(-(&dv.x)[u]);
        prow[j + u] = (__bf16)e;
      }
    }
  }
  __syncthreads();

  // ---- pass 3: O = P @ V  (A-frags from LDS, B-frags contiguous from V^T)
  v8f co[4] = {};
  const __bf16* vt = VTb + bh*DHEAD*SEQ;
  auto ldv = [&](int j0, v16bf* vb) {
    #pragma unroll
    for (int f = 0; f < 4; ++f) {
      const __bf16* bp = vt + (size_t)(f*16 + lr)*SEQ + j0 + lh*16;
      vb[f] = ld_frag(bp, bp + 8);
    }
  };
  auto pv_body = [&](int j0, const v16bf* vb) {
    const __bf16* ap = P + lr*SEQ + j0 + lh*16;
    v16bf a = ld_frag(ap, ap + 8);
    #pragma unroll
    for (int f = 0; f < 4; ++f) co[f] = WMMA_BF16(a, vb[f], co[f]);
  };
  v16bf vb[4];
  ldv(0, vb);
  for (int j0 = 0; j0 < SEQ - 32; j0 += 32) {
    v16bf nv[4];
    ldv(j0 + 32, nv);                   // in flight across current WMMAs
    pv_body(j0, vb);
    #pragma unroll
    for (int f = 0; f < 4; ++f) vb[f] = nv[f];
  }
  pv_body(SEQ - 32, vb);

  #pragma unroll
  for (int f = 0; f < 4; ++f) {
    #pragma unroll
    for (int r = 0; r < 8; ++r) {
      const int row = i0 + lh*8 + r;
      const int col = h*DHEAD + f*16 + lr;
      Ob[((size_t)b*SEQ + row)*INNER + col] = (__bf16)co[f][r];
    }
  }
}

// ---------------------------------------------------------------- out proj
__global__ __launch_bounds__(256) void out_gemm_kernel(
    const __bf16* __restrict__ Ob, const __bf16* __restrict__ WoutT,
    const float* __restrict__ bias, float* __restrict__ out) {
  const int lane = threadIdx.x & 31, wave = threadIdx.x >> 5;
  const int lr = lane & 15, lh = lane >> 4;
  const int row0 = blockIdx.x * 128 + wave * 16;
  const int col0 = blockIdx.y * 64;

  v8f acc[4] = {};
  const __bf16* arow = Ob + (size_t)(row0 + lr)*INNER + lh*16;
  const __bf16* bp0 = WoutT + (size_t)(col0 +  0 + lr)*DMODEL + lh*16;
  const __bf16* bp1 = WoutT + (size_t)(col0 + 16 + lr)*DMODEL + lh*16;
  const __bf16* bp2 = WoutT + (size_t)(col0 + 32 + lr)*DMODEL + lh*16;
  const __bf16* bp3 = WoutT + (size_t)(col0 + 48 + lr)*DMODEL + lh*16;

  FragSet cur = load_set(arow, bp0, bp1, bp2, bp3, 0);
  for (int k0 = 0; k0 < DMODEL - 32; k0 += 32) {
    FragSet nxt = load_set(arow, bp0, bp1, bp2, bp3, k0 + 32);
    mma_set(acc, cur);
    cur = nxt;
  }
  mma_set(acc, cur);

  #pragma unroll
  for (int f = 0; f < 4; ++f) {
    #pragma unroll
    for (int r = 0; r < 8; ++r) {
      const int grow = row0 + lh*8 + r;
      const int col  = col0 + f*16 + lr;
      out[(size_t)grow*DMODEL + col] = acc[f][r] + bias[col];
    }
  }
}

// ---------------------------------------------------------------- launch
extern "C" void kernel_launch(void* const* d_in, const int* in_sizes, int n_in,
                              void* d_out, int out_size, void* d_ws, size_t ws_size,
                              hipStream_t stream) {
  (void)in_sizes; (void)n_in; (void)out_size; (void)ws_size;
  const float*         x    = (const float*)d_in[0];
  const unsigned char* mask = (const unsigned char*)d_in[1];  // bool array
  const float*         adj  = (const float*)d_in[2];
  const float*         dist = (const float*)d_in[3];
  const float*         wqkv = (const float*)d_in[4];
  const float*         wout = (const float*)d_in[5];
  const float*         bias = (const float*)d_in[6];

  char* ws = (char*)d_ws;                       // ~22 MB used, all 16B-aligned
  __bf16* Xb    = (__bf16*)(ws);                //  4 MB  [4096,512]
  __bf16* WqkvT = (__bf16*)(ws +  4194304);     //  1.5MB [1536,512]
  __bf16* WoutT = (__bf16*)(ws +  5767168);     //  0.5MB [512,512]
  __bf16* Qb    = (__bf16*)(ws +  6291456);     //  4 MB  [B,H,N,Dh] (pre-scaled)
  __bf16* Kb    = (__bf16*)(ws + 10485760);     //  4 MB  [B,H,N,Dh]
  __bf16* VTb   = (__bf16*)(ws + 14680064);     //  4 MB  [B,H,Dh,N]
  __bf16* Ob    = (__bf16*)(ws + 18874368);     //  4 MB  [B*N, 512]

  (void)hipFuncSetAttribute((const void*)attn_kernel,
                            hipFuncAttributeMaxDynamicSharedMemorySize, SMEM_BYTES);

  cvt_kernel<<<12288, 256, 0, stream>>>(x, wqkv, wout, Xb, WqkvT, WoutT);
  qkv_gemm_kernel<<<dim3(32, 24), 256, 0, stream>>>(Xb, WqkvT, Qb, Kb, VTb);
  attn_kernel<<<BATCH*HEADS*(SEQ/128), 256, SMEM_BYTES, stream>>>(
      mask, adj, dist, Qb, Kb, VTb, Ob);
  out_gemm_kernel<<<dim3(32, 8), 256, 0, stream>>>(Ob, WoutT, bias, (float*)d_out);
}
